// MultiHeadedSelfAttentionModule_83107617177635
// MI455X (gfx1250) — matrix operand order repaired and checked by reading, not compiled, not run on hardware
//
#include <hip/hip_runtime.h>
#include <hip/hip_bf16.h>

// ---------------------------------------------------------------------------
// CDNA5 (gfx1250) wave32 WMMA implementation of relative-position MHSA.
// LDS tiles kept in fragment-native layout (2x ds_load_b128 per fragment);
// GEMM is double-buffered and register-pipelined.
// ---------------------------------------------------------------------------

typedef __attribute__((ext_vector_type(16))) __bf16 v16bf;
typedef __attribute__((ext_vector_type(4)))  __bf16 v4bf;
typedef __attribute__((ext_vector_type(8)))  float  v8f;

#define WMMA_BF16(a, b, c) \
    __builtin_amdgcn_wmma_f32_16x16x32_bf16(false, (a), false, (b), (short)0, (c), false, false)

constexpr int B_  = 4;
constexpr int L_  = 1024;
constexpr int D_  = 1024;
constexpr int H_  = 16;
constexpr int DH_ = 64;
constexpr int R_  = 2 * L_ - 1;   // 2047 relative positions

// Inverse fragment maps (CDNA5 ISA 7.12.2, wave32):
// A 16x32 bf16: element (m,k) -> lane m+16*((k>>3)&1), elem (k&7)+8*((k>>4)&1)
__device__ __forceinline__ int a_lane(int m, int k) { return m + 16 * ((k >> 3) & 1); }
__device__ __forceinline__ int a_elem(int k)        { return (k & 7) + 8 * ((k >> 4) & 1); }
// B 32x16 bf16: element (k,n) -> lane n+16*((k>>3)&1), elem (k&7)+8*((k>>4)&1)
__device__ __forceinline__ int b_lane(int n, int k) { return n + 16 * ((k >> 3) & 1); }
__device__ __forceinline__ int b_elem(int k)        { return (k & 7) + 8 * ((k >> 4) & 1); }

__device__ __forceinline__ void st4(v16bf* fragBase, int lane, int elem, v4bf val) {
    *(v4bf*)((__bf16*)&fragBase[lane] + elem) = val;
}
__device__ __forceinline__ v4bf cvt4(float4 f) {
    return (v4bf){(__bf16)f.x, (__bf16)f.y, (__bf16)f.z, (__bf16)f.w};
}

// ---------------------------------------------------------------------------
// Kernel 1: elementwise affine x = scale*in + shift
// ---------------------------------------------------------------------------
__global__ __launch_bounds__(256) void affine_kernel(
    const float* __restrict__ in, const float* __restrict__ scale,
    const float* __restrict__ shift, float* __restrict__ x, int total, int D)
{
    int i = blockIdx.x * 256 + threadIdx.x;
    if (i < total) {
        int d = i & (D - 1);
        x[i] = scale[d] * in[i] + shift[d];
    }
}

// ---------------------------------------------------------------------------
// Kernel 2: sinusoidal relative position table, positions -(L-1)..(L-1)
// ---------------------------------------------------------------------------
__global__ __launch_bounds__(256) void posemb_kernel(float* __restrict__ pe,
                                                     int Rr, int D, int L)
{
    int i = blockIdx.x * 256 + threadIdx.x;
    if (i < Rr * D) {
        int r = i / D, d = i - r * D;
        float pos = (float)(r - (L - 1));
        float invf = __expf(-(float)(d & ~1) * (9.210340371976184f / (float)D));
        float ang  = pos * invf;
        pe[i] = (d & 1) ? cosf(ang) : sinf(ang);
    }
}

// ---------------------------------------------------------------------------
// Kernel 3: tiled WMMA bf16 GEMM  C[M,N] = A[M,K] @ W[K,N] (+bias)(+residual)
// 128 threads = 4 waves; 128x64 C tile; K step 32; double-buffered LDS with
// register-staged global loads (software pipelining, no divergent guards).
// ---------------------------------------------------------------------------
__global__ __launch_bounds__(128) void gemm_wmma_kernel(
    const float* __restrict__ A, const float* __restrict__ W,
    const float* __restrict__ bias, const float* __restrict__ residual,
    float* __restrict__ C, int M, int N, int K)
{
    __shared__ v16bf sAf[2][8][32];   // [buf][row-tile f][lane]  A frags (16x32)
    __shared__ v16bf sBf[2][4][32];   // [buf][n-group ng][lane]  B frags (32x16)

    const int tid  = threadIdx.x;
    const int wave = tid >> 5;
    const int lane = tid & 31;
    const int m0   = blockIdx.x * 128;
    const int n0   = blockIdx.y * 64;

    // per-thread staging coordinates
    const int arow = tid;                 // A row 0..127 (one row per thread)
    const int k4   = (tid >> 4) * 4;      // B k base (0..28)
    const int n4   = (tid & 15) * 4;      // B n base (0..60)
    int arowc = m0 + arow;                // clamp instead of branch: OOB A rows
    if (arowc > M - 1) arowc = M - 1;     // only feed OOB C rows (never stored)

    float4 aReg[8];
    float4 bReg[4];

    auto load_tile = [&](int k0) {
        const float* ap = &A[(size_t)arowc * K + k0];
#pragma unroll
        for (int i = 0; i < 8; ++i) aReg[i] = *(const float4*)(ap + 4 * i);
#pragma unroll
        for (int i = 0; i < 4; ++i)
            bReg[i] = *(const float4*)&W[(size_t)(k0 + k4 + i) * N + n0 + n4];
    };
    auto store_tile = [&](int buf) {
        const int f = arow >> 4, m = arow & 15;
#pragma unroll
        for (int i = 0; i < 8; ++i) {
            int kk = 4 * i;
            st4(sAf[buf][f], a_lane(m, kk), a_elem(kk), cvt4(aReg[i]));
        }
        const float* c0 = &bReg[0].x; const float* c1 = &bReg[1].x;
        const float* c2 = &bReg[2].x; const float* c3 = &bReg[3].x;
#pragma unroll
        for (int nn = 0; nn < 4; ++nn) {
            int nc = n4 + nn;
            v4bf kq = (v4bf){(__bf16)c0[nn], (__bf16)c1[nn], (__bf16)c2[nn], (__bf16)c3[nn]};
            st4(sBf[buf][nc >> 4], b_lane(nc & 15, k4), b_elem(k4), kq);
        }
    };

    v8f acc[2][4] = {};

    load_tile(0);
    store_tile(0);
    const int nk = K >> 5;
    for (int t = 0; t < nk; ++t) {
        const int cur = t & 1;
        if (t + 1 < nk) load_tile((t + 1) << 5);   // global loads overlap compute
        __syncthreads();
        v16bf a0 = sAf[cur][wave * 2 + 0][lane];
        v16bf a1 = sAf[cur][wave * 2 + 1][lane];
#pragma unroll
        for (int ng = 0; ng < 4; ++ng) {
            v16bf bf_ = sBf[cur][ng][lane];
            acc[0][ng] = WMMA_BF16(a0, bf_, acc[0][ng]);
            acc[1][ng] = WMMA_BF16(a1, bf_, acc[1][ng]);
        }
        if (t + 1 < nk) store_tile(cur ^ 1);
    }

    const int n  = lane & 15;
    const int mh = (lane >> 4) * 8;
#pragma unroll
    for (int f = 0; f < 2; ++f)
#pragma unroll
        for (int ng = 0; ng < 4; ++ng)
#pragma unroll
            for (int r = 0; r < 8; ++r) {
                int row = m0 + wave * 32 + f * 16 + mh + r;
                int col = n0 + ng * 16 + n;
                if (row < M) {
                    float val = acc[f][ng][r];
                    if (bias)     val += bias[col];
                    if (residual) val += residual[(size_t)row * N + col];
                    C[(size_t)row * N + col] = val;
                }
            }
}

// ---------------------------------------------------------------------------
// Kernel 4: flash-style attention with relative-position band WMMA.
// Grid: (L/64, H, B); 128 threads = 4 waves, wave w owns query rows i0+16w..+15.
// ---------------------------------------------------------------------------
__global__ __launch_bounds__(128) void attn_kernel(
    const float* __restrict__ q, const float* __restrict__ k,
    const float* __restrict__ v, const float* __restrict__ p,
    const float* __restrict__ ubias, const float* __restrict__ vbias,
    const unsigned char* __restrict__ mask, float* __restrict__ ctx)
{
    const int b    = blockIdx.z;
    const int h    = blockIdx.y;
    const int i0   = blockIdx.x * 64;
    const int tid  = threadIdx.x;
    const int wave = tid >> 5;
    const int lane = tid & 31;

    __shared__ v16bf sQu[4][2][32];   // [row-tile][kg][lane]  (q+u) A-frags
    __shared__ v16bf sQv[4][2][32];   // (q+v) A-frags
    __shared__ v16bf sKf[2][2][32];   // [key ng][kg][lane]    K^T B-frags (n=key, k=d&31)
    __shared__ v16bf sVf[4][32];      // [d ng][lane]          V  B-frags (k=key, n=d&15)
    __shared__ v16bf sPf[6][2][32];   // [band rg][kg][lane]   P^T B-frags
    __shared__ float sGat[4][16][48]; // per-wave pos-score gather buffer
    __shared__ v16bf sPij[4][32];     // per-wave softmax-prob A-frag
    __shared__ unsigned char sMask[L_];

    // ---- stage Q(+u), Q(+v) as A-fragments ----
    {
        int row  = tid >> 1;          // 0..63
        int dblk = (tid & 1) * 32;    // kg half
        int f    = row >> 4;
        int m    = row & 15;
        int kg   = dblk >> 5;
        const float* qp = &q[(((size_t)b * L_ + i0 + row) * H_ + h) * DH_ + dblk];
        const float* up = &ubias[h * DH_ + dblk];
        const float* vp = &vbias[h * DH_ + dblk];
#pragma unroll
        for (int i = 0; i < 8; ++i) {
            int kk = 4 * i;           // within the 32-wide half
            float4 q4 = *(const float4*)(qp + kk);
            float4 u4 = *(const float4*)(up + kk);
            float4 w4 = *(const float4*)(vp + kk);
            v4bf qu = (v4bf){(__bf16)(q4.x + u4.x), (__bf16)(q4.y + u4.y),
                             (__bf16)(q4.z + u4.z), (__bf16)(q4.w + u4.w)};
            v4bf qv = (v4bf){(__bf16)(q4.x + w4.x), (__bf16)(q4.y + w4.y),
                             (__bf16)(q4.z + w4.z), (__bf16)(q4.w + w4.w)};
            st4(sQu[f][kg], a_lane(m, kk), a_elem(kk), qu);
            st4(sQv[f][kg], a_lane(m, kk), a_elem(kk), qv);
        }
    }
    for (int j = tid; j < L_; j += 128) sMask[j] = mask[(size_t)b * L_ + j];
    __syncthreads();

    v16bf aU[2], aV[2];
    aU[0] = sQu[wave][0][lane]; aU[1] = sQu[wave][1][lane];
    aV[0] = sQv[wave][0][lane]; aV[1] = sQv[wave][1][lane];

    v8f o[4] = {};
    float mrun[8], lrun[8];
#pragma unroll
    for (int r = 0; r < 8; ++r) { mrun[r] = -3.0e38f; lrun[r] = 0.f; }

    const float sscale = 0.03125f;    // 1/sqrt(1024)
    const int n  = lane & 15;
    const int mh = (lane >> 4) * 8;

    for (int j0 = 0; j0 < L_; j0 += 32) {
        __syncthreads();
        // ---- stage K tile (32x64) as K^T B-frags ----
        {
            int jj   = tid >> 2;              // key row 0..31
            int dblk = (tid & 3) * 16;
            const float* kp = &k[(((size_t)b * L_ + j0 + jj) * H_ + h) * DH_ + dblk];
#pragma unroll
            for (int i = 0; i < 4; ++i) {
                int d  = dblk + 4 * i;
                int kg = d >> 5, kk = d & 31;
                float4 k4 = *(const float4*)(kp + 4 * i);
                st4(sKf[jj >> 4][kg], b_lane(jj & 15, kk), b_elem(kk), cvt4(k4));
            }
        }
        // ---- stage V tile (32x64) as B-frags (transpose 4x4 in regs) ----
        {
            int jb = (tid >> 4) * 4;          // key base 0..28
            int d  = (tid & 15) * 4;          // col base
            float4 r0 = *(const float4*)&v[(((size_t)b * L_ + j0 + jb + 0) * H_ + h) * DH_ + d];
            float4 r1 = *(const float4*)&v[(((size_t)b * L_ + j0 + jb + 1) * H_ + h) * DH_ + d];
            float4 r2 = *(const float4*)&v[(((size_t)b * L_ + j0 + jb + 2) * H_ + h) * DH_ + d];
            float4 r3 = *(const float4*)&v[(((size_t)b * L_ + j0 + jb + 3) * H_ + h) * DH_ + d];
            const float* c0 = &r0.x; const float* c1 = &r1.x;
            const float* c2 = &r2.x; const float* c3 = &r3.x;
#pragma unroll
            for (int dd = 0; dd < 4; ++dd) {
                int dc = d + dd;
                v4bf vq = (v4bf){(__bf16)c0[dd], (__bf16)c1[dd], (__bf16)c2[dd], (__bf16)c3[dd]};
                st4(sVf[dc >> 4], b_lane(dc & 15, jb), b_elem(jb), vq);
            }
        }
        // ---- stage position band (96 rows) as P^T B-frags ----
        const int rbase = j0 - i0 + (L_ - 1) - 63;
        for (int it = tid; it < 96 * 16; it += 128) {
            int rr   = it >> 4;               // band row 0..95
            int dcol = (it & 15) * 4;
            int rg   = rr >> 4;
            int kg   = dcol >> 5, kk = dcol & 31;
            int rglob = rbase + rr;
            rglob = rglob < 0 ? 0 : (rglob > R_ - 1 ? R_ - 1 : rglob);
            float4 p4 = *(const float4*)&p[((size_t)rglob * H_ + h) * DH_ + dcol];
            st4(sPf[rg][kg], b_lane(rr & 15, kk), b_elem(kk), cvt4(p4));
        }
        __syncthreads();

        // ---- content scores: (q+u) @ K^T ----
        v8f sc[2] = {};
#pragma unroll
        for (int ng = 0; ng < 2; ++ng)
#pragma unroll
            for (int kg = 0; kg < 2; ++kg) {
                v16bf bk = sKf[ng][kg][lane];
                sc[ng] = WMMA_BF16(aU[kg], bk, sc[ng]);
            }

        // ---- positional band scores: (q+v) @ Pband^T (wave band = rg 3-w ..) ----
        v8f pc[3] = {};
#pragma unroll
        for (int ng = 0; ng < 3; ++ng)
#pragma unroll
            for (int kg = 0; kg < 2; ++kg) {
                v16bf bp = sPf[3 - wave + ng][kg][lane];
                pc[ng] = WMMA_BF16(aV[kg], bp, pc[ng]);
            }

        // spill pos tile for the diagonal (rel-shift) gather
#pragma unroll
        for (int ng = 0; ng < 3; ++ng)
#pragma unroll
            for (int r = 0; r < 8; ++r)
                sGat[wave][mh + r][ng * 16 + n] = pc[ng][r];
        asm volatile("s_wait_dscnt 0" ::: "memory");

        // ---- gather pos term, mask, scale; tile row max ----
        float sv[2][8];
        float tmax[8];
#pragma unroll
        for (int r = 0; r < 8; ++r) tmax[r] = -3.0e38f;
#pragma unroll
        for (int ng = 0; ng < 2; ++ng)
#pragma unroll
            for (int r = 0; r < 8; ++r) {
                int m  = mh + r;
                int jl = ng * 16 + n;
                float val = (sc[ng][r] + sGat[wave][m][15 + jl - m]) * sscale;
                if (!sMask[j0 + jl]) val = -1.0e9f;
                sv[ng][r] = val;
                tmax[r] = fmaxf(tmax[r], val);
            }
#pragma unroll
        for (int r = 0; r < 8; ++r) {
            float t = tmax[r];
            t = fmaxf(t, __shfl_xor(t, 1, 32));
            t = fmaxf(t, __shfl_xor(t, 2, 32));
            t = fmaxf(t, __shfl_xor(t, 4, 32));
            t = fmaxf(t, __shfl_xor(t, 8, 32));
            tmax[r] = t;
        }

        // ---- online softmax update ----
        float corr[8];
#pragma unroll
        for (int r = 0; r < 8; ++r) {
            float mnew = fmaxf(mrun[r], tmax[r]);
            corr[r] = __expf(mrun[r] - mnew);
            mrun[r] = mnew;
            float e0 = __expf(sv[0][r] - mnew);
            float e1 = __expf(sv[1][r] - mnew);
            sv[0][r] = e0; sv[1][r] = e1;
            float ts = e0 + e1;
            ts += __shfl_xor(ts, 1, 32);
            ts += __shfl_xor(ts, 2, 32);
            ts += __shfl_xor(ts, 4, 32);
            ts += __shfl_xor(ts, 8, 32);
            lrun[r] = lrun[r] * corr[r] + ts;
        }
#pragma unroll
        for (int ng = 0; ng < 4; ++ng)
#pragma unroll
            for (int r = 0; r < 8; ++r) o[ng][r] *= corr[r];

        // ---- P (16x32) -> A-fragment via LDS, then o += P @ V ----
#pragma unroll
        for (int ng = 0; ng < 2; ++ng)
#pragma unroll
            for (int r = 0; r < 8; ++r) {
                int m = mh + r, kcol = ng * 16 + n;
                ((__bf16*)&sPij[wave][a_lane(m, kcol)])[a_elem(kcol)] = (__bf16)sv[ng][r];
            }
        asm volatile("s_wait_dscnt 0" ::: "memory");

        v16bf ap = sPij[wave][lane];
#pragma unroll
        for (int ng = 0; ng < 4; ++ng) {
            v16bf bv_ = sVf[ng][lane];
            o[ng] = WMMA_BF16(ap, bv_, o[ng]);
        }
    }

    // ---- normalize and write ctx[b, i, h, d] ----
#pragma unroll
    for (int ng = 0; ng < 4; ++ng)
#pragma unroll
        for (int r = 0; r < 8; ++r) {
            int row = i0 + wave * 16 + mh + r;
            float inv = 1.0f / lrun[r];
            ctx[(((size_t)b * L_ + row) * H_ + h) * DH_ + ng * 16 + n] = o[ng][r] * inv;
        }
}

// ---------------------------------------------------------------------------
// Host-side launch
// ---------------------------------------------------------------------------
extern "C" void kernel_launch(void* const* d_in, const int* in_sizes, int n_in,
                              void* d_out, int out_size, void* d_ws, size_t ws_size,
                              hipStream_t stream) {
    (void)in_sizes; (void)n_in; (void)out_size; (void)ws_size;

    const float*         inputs = (const float*)d_in[0];
    const unsigned char* mask   = (const unsigned char*)d_in[1];
    const float*         scale  = (const float*)d_in[2];
    const float*         shift  = (const float*)d_in[3];
    const float*         Wq     = (const float*)d_in[4];
    const float*         bq     = (const float*)d_in[5];
    const float*         Wk     = (const float*)d_in[6];
    const float*         bk     = (const float*)d_in[7];
    const float*         Wv     = (const float*)d_in[8];
    const float*         bv     = (const float*)d_in[9];
    const float*         Wpos   = (const float*)d_in[10];
    const float*         ub     = (const float*)d_in[11];
    const float*         vb     = (const float*)d_in[12];
    const float*         Wo     = (const float*)d_in[13];
    const float*         bo     = (const float*)d_in[14];
    float*               out    = (float*)d_out;

    const size_t BL = (size_t)B_ * L_;

    char* ws = (char*)d_ws;
    float* x   = (float*)ws; ws += BL * D_ * sizeof(float);
    float* pe  = (float*)ws; ws += (size_t)R_ * D_ * sizeof(float);
    float* q   = (float*)ws; ws += BL * D_ * sizeof(float);
    float* k   = (float*)ws; ws += BL * D_ * sizeof(float);
    float* v   = (float*)ws; ws += BL * D_ * sizeof(float);
    float* p   = (float*)ws; ws += (size_t)R_ * D_ * sizeof(float);
    float* ctx = (float*)ws; ws += BL * D_ * sizeof(float);

    {
        int total = (int)(BL * D_);
        affine_kernel<<<(total + 255) / 256, 256, 0, stream>>>(inputs, scale, shift, x, total, D_);
    }
    {
        int total = R_ * D_;
        posemb_kernel<<<(total + 255) / 256, 256, 0, stream>>>(pe, R_, D_, L_);
    }
    dim3 gQ((unsigned)((BL + 127) / 128), D_ / 64);
    dim3 gP((unsigned)((R_ + 127) / 128), D_ / 64);
    gemm_wmma_kernel<<<gQ, 128, 0, stream>>>(x,  Wq,   bq,      nullptr, q, (int)BL, D_, D_);
    gemm_wmma_kernel<<<gQ, 128, 0, stream>>>(x,  Wk,   bk,      nullptr, k, (int)BL, D_, D_);
    gemm_wmma_kernel<<<gQ, 128, 0, stream>>>(x,  Wv,   bv,      nullptr, v, (int)BL, D_, D_);
    gemm_wmma_kernel<<<gP, 128, 0, stream>>>(pe, Wpos, nullptr, nullptr, p, R_,      D_, D_);
    {
        dim3 gA(L_ / 64, H_, B_);
        attn_kernel<<<gA, 128, 0, stream>>>(q, k, v, p, ub, vb, mask, ctx);
    }
    gemm_wmma_kernel<<<gQ, 128, 0, stream>>>(ctx, Wo, bo, inputs, out, (int)BL, D_, D_);
}
